// TSERMamba_6811818131839
// MI455X (gfx1250) — matrix operand order repaired and checked by reading, not compiled, not run on hardware
//
#include <hip/hip_runtime.h>
#include <math.h>

// ---------------- WMMA types ----------------
typedef __attribute__((ext_vector_type(16))) __bf16 v16bf;
typedef __attribute__((ext_vector_type(4)))  __bf16 v4bf;
typedef __attribute__((ext_vector_type(8)))  float  v8f;

// Problem constants (from reference)
#define BB 32
#define CC 32
#define PS 256
#define IM 128
#define PP 16
#define SL 1024
#define DSN 16
#define DCN 4
#define D1 (3*PS)         // 768
#define NP 64             // (IM/P)^2

__device__ __forceinline__ float siluf(float x){ return x / (1.f + __expf(-x)); }
__device__ __forceinline__ v4bf cvt4(float4 f){
  return (v4bf){(__bf16)f.x, (__bf16)f.y, (__bf16)f.z, (__bf16)f.w};
}

// ---------------- X loaders for GEMM ----------------
// load4 contract (FULL path only): k % 4 == 0, row is k-contiguous & 16B aligned.
struct DenseLoader {
  const float* p; long ld;   // ld % 4 == 0 for all FULL-path uses
  __device__ float  load (int n, int k) const { return p[(long)n*ld + k]; }
  __device__ float4 load4(int n, int k) const { return *(const float4*)(p + (long)n*ld + k); }
};
// im2col of x_cwt for patch conv: n=(b,ph,pw), k=(ci,kh,kw)
struct PatchLoader {
  const float* x;
  __device__ float load(int n, int k) const {
    int b = n >> 6, ph = (n >> 3) & 7, pw = n & 7;
    int ci = k >> 8, kh = (k >> 4) & 15, kw = k & 15;
    return x[(((long)(b*CC + ci)*IM + ph*PP + kh)*IM) + pw*PP + kw];
  }
  __device__ float4 load4(int n, int k) const {   // kw%4==0; quad stays inside 16-wide run
    int b = n >> 6, ph = (n >> 3) & 7, pw = n & 7;
    int ci = k >> 8, kh = (k >> 4) & 15, kw = k & 15;
    return *(const float4*)&x[(((long)(b*CC + ci)*IM + ph*PP + kh)*IM) + pw*PP + kw];
  }
};
// Gather xp_buf[(b*64+p)*32 + c] for the patch projection: n=(b,c), k=p
struct PatchProjLoader {
  const float* xp;
  __device__ float load(int n, int k) const {
    int b = n >> 5, c = n & 31;
    return xp[((long)(b*NP + k))*CC + c];
  }
  __device__ float4 load4(int n, int k) const {   // strided gather; 4 independent b32 loads
    float4 r; r.x = load(n, k); r.y = load(n, k+1); r.z = load(n, k+2); r.w = load(n, k+3);
    return r;
  }
};

// ---------------- bf16 WMMA GEMM (FULL path) ----------------
// out[n, m] = sum_k X[n,k] * W[m,k] (+bias[m]) (+add[n,m]) (act: 0 none, 1 softplus)
// Requires N%64==0, K%32==0, M%(MI*16)==0. Block = 128 threads (4 waves).
// Tile: 64 rows (n) x MI*16 cols (m); wave w owns n-subtile w; MI straight-line WMMAs
// per k-chunk share one B fragment. LDS double buffer; staging = batched b128 loads
// into registers, then paired bf16 converts + ds_store_b64.
template <class XL, int MI>
__global__ void gemm_wmma_full(XL xl, const float* __restrict__ Wm,
                               const float* __restrict__ bias, const float* __restrict__ add,
                               float* __restrict__ out, int N, int K, int M, int ldOut, int act)
{
  __shared__ __bf16 As[2][MI*16][32];   // [buf][m][k]
  __shared__ __bf16 Bs[2][64][32];      // [buf][n][k]
  const int tid  = threadIdx.x;
  const int lane = tid & 31, wave = tid >> 5;
  const int half = lane >> 4, mn = lane & 15;
  const int nBase = blockIdx.x * 64;
  const int mBase = blockIdx.y * (MI*16);
  const int nk = K >> 5;
  v8f acc[MI] = {};

  auto stage = [&](int buf, int k0) {
    // phase 1: batched 128-bit loads into registers (single clause, one wait)
    float4 ta[MI], tb[4];
#pragma unroll
    for (int i = 0; i < MI; ++i) {
      int q = tid + 128*i;                 // quad index over MI*16 x 32 A tile
      int m = q >> 3, kq = (q & 7) << 2;
      const float* ap = &Wm[(long)(mBase + m)*K + k0 + kq];
      ta[i] = *(const float4*)ap;
      if (i == 0) __builtin_prefetch(ap + 64, 0, 3);   // chunk after next (A)
    }
#pragma unroll
    for (int i = 0; i < 4; ++i) {
      int q = tid + 128*i;                 // quad index over 64 x 32 B tile
      int n = q >> 3, kq = (q & 7) << 2;
      tb[i] = xl.load4(nBase + n, k0 + kq);
    }
    // phase 2: paired converts + 64-bit LDS stores
#pragma unroll
    for (int i = 0; i < MI; ++i) {
      int q = tid + 128*i;
      int m = q >> 3, kq = (q & 7) << 2;
      *(v4bf*)&As[buf][m][kq] = cvt4(ta[i]);
    }
#pragma unroll
    for (int i = 0; i < 4; ++i) {
      int q = tid + 128*i;
      int n = q >> 3, kq = (q & 7) << 2;
      *(v4bf*)&Bs[buf][n][kq] = cvt4(tb[i]);
    }
  };

  stage(0, 0);
  __syncthreads();
  for (int ks = 0; ks < nk; ++ks) {
    const int cur = ks & 1;
    if (ks + 1 < nk) stage(cur ^ 1, (ks + 1) << 5);    // overlap next-chunk loads with WMMA
    v16bf bfrag;
#pragma unroll
    for (int e = 0; e < 16; ++e) bfrag[e] = Bs[cur][wave*16 + mn][e + 16*half];
#pragma unroll
    for (int mi = 0; mi < MI; ++mi) {                  // straight-line, no branches
      v16bf afrag;
#pragma unroll
      for (int e = 0; e < 16; ++e) {
        int ka = e + 16*half;
        afrag[e] = As[cur][mi*16 + mn][ka];
      }
      acc[mi] = __builtin_amdgcn_wmma_f32_16x16x32_bf16(false, afrag, false, bfrag,
                                                        (short)0, acc[mi], false, false);
    }
    __syncthreads();
  }
  // epilogue (no guards: all dims exact multiples)
  const int n = nBase + wave*16 + mn;
#pragma unroll
  for (int mi = 0; mi < MI; ++mi) {
#pragma unroll
    for (int r = 0; r < 8; ++r) {
      int m = mBase + mi*16 + r + 8*half;
      float v = acc[mi][r];
      if (bias) v += bias[m];
      if (add)  v += add[(long)n*ldOut + m];
      if (act == 1) v = (v > 20.f) ? v : log1pf(__expf(v));
      out[(long)n*ldOut + m] = v;
    }
  }
}

// ---------------- bf16 WMMA GEMM (generic guarded path, small/odd shapes) ----------------
template <class XL>
__global__ void gemm_wmma_gen(XL xl, const float* __restrict__ Wm,
                              const float* __restrict__ bias, const float* __restrict__ add,
                              float* __restrict__ out, int N, int K, int M, int ldOut, int act)
{
  __shared__ __bf16 As[64][32];
  __shared__ __bf16 Bs[64][32];
  const int tid  = threadIdx.x;
  const int lane = tid & 31, wave = tid >> 5;
  const int half = lane >> 4, mn = lane & 15;
  const int nBase = blockIdx.x * 64;
  const int mBase = blockIdx.y * 64;
  v8f acc[4] = {};

  for (int k0 = 0; k0 < K; k0 += 32) {
#pragma unroll
    for (int i = 0; i < 16; ++i) {
      int idx = tid + 128*i;
      int m = idx >> 5, k = idx & 31;
      float va = 0.f, vb = 0.f;
      if (mBase + m < M && k0 + k < K) va = Wm[(long)(mBase + m)*K + k0 + k];
      if (nBase + m < N && k0 + k < K) vb = xl.load(nBase + m, k0 + k);
      As[m][k] = (__bf16)va;
      Bs[m][k] = (__bf16)vb;
    }
    __syncthreads();
    v16bf bfrag;
#pragma unroll
    for (int e = 0; e < 16; ++e) bfrag[e] = Bs[wave*16 + mn][e + 16*half];
#pragma unroll
    for (int mi = 0; mi < 4; ++mi) {
      v16bf afrag;
#pragma unroll
      for (int e = 0; e < 16; ++e) {
        int ka = e + 16*half;
        afrag[e] = As[mi*16 + mn][ka];
      }
      acc[mi] = __builtin_amdgcn_wmma_f32_16x16x32_bf16(false, afrag, false, bfrag,
                                                        (short)0, acc[mi], false, false);
    }
    __syncthreads();
  }
  const int n = nBase + wave*16 + mn;
#pragma unroll
  for (int mi = 0; mi < 4; ++mi) {
#pragma unroll
    for (int r = 0; r < 8; ++r) {
      int m = mBase + mi*16 + r + 8*half;
      if (m < M && n < N) {
        float v = acc[mi][r];
        if (bias) v += bias[m];
        if (add)  v += add[(long)n*ldOut + m];
        if (act == 1) v = (v > 20.f) ? v : log1pf(__expf(v));
        out[(long)n*ldOut + m] = v;
      }
    }
  }
}

template <class XL>
static void gemm_launch(XL xl, const float* Wm, const float* bias, const float* add,
                        float* out, int N, int K, int M, int ldOut, int act, hipStream_t s)
{
  const bool fullNK = (N % 64 == 0) && (K % 32 == 0);
  if (fullNK && (M % 64 == 0)) {
    dim3 g(N/64, M/64);
    gemm_wmma_full<XL, 4><<<g, dim3(128), 0, s>>>(xl, Wm, bias, add, out, N, K, M, ldOut, act);
  } else if (fullNK && (M % 32 == 0)) {
    dim3 g(N/64, M/32);
    gemm_wmma_full<XL, 2><<<g, dim3(128), 0, s>>>(xl, Wm, bias, add, out, N, K, M, ldOut, act);
  } else if (fullNK && (M % 16 == 0)) {
    dim3 g(N/64, M/16);
    gemm_wmma_full<XL, 1><<<g, dim3(128), 0, s>>>(xl, Wm, bias, add, out, N, K, M, ldOut, act);
  } else {
    dim3 g((N + 63)/64, (M + 63)/64);
    gemm_wmma_gen<XL><<<g, dim3(128), 0, s>>>(xl, Wm, bias, add, out, N, K, M, ldOut, act);
  }
}

// ---------------- elementwise kernels ----------------
__global__ void copy_kernel(float* dst, const float* src, long n){
  long i = (long)blockIdx.x*256 + threadIdx.x;
  if (i < n) dst[i] = src[i];
}
__global__ void add_kernel(float* dst, const float* src, long n){
  long i = (long)blockIdx.x*256 + threadIdx.x;
  if (i < n) dst[i] += src[i];
}
// dst[(b,l,d)] = src[(b, L-1-l, d)]
__global__ void flip_seq_kernel(float* dst, const float* src, int Bn, int L, int D){
  long tot = (long)Bn*L*D;
  long i = (long)blockIdx.x*256 + threadIdx.x;
  if (i >= tot) return;
  int d = (int)(i % D); long r = i / D;
  int l = (int)(r % L); int b = (int)(r / L);
  dst[i] = src[((long)(b*L + (L-1-l)))*D + d];
}
// src (B,L,D) -> dst (B,D,L)
__global__ void transpose_bld_kernel(float* dst, const float* src, int Bn, int L, int D){
  long tot = (long)Bn*L*D;
  long i = (long)blockIdx.x*256 + threadIdx.x;
  if (i >= tot) return;
  int l = (int)(i % L); long r = i / L;
  int d = (int)(r % D); int b = (int)(r / D);
  dst[i] = src[((long)b*L + l)*D + d];
}
// catpre[n*768 + 512 + j] = x_features[n*1280 + j]
__global__ void copy_feat_kernel(float* catpre, const float* xf){
  int i = blockIdx.x*256 + threadIdx.x; // 1024*128
  if (i >= 1024*128) return;
  int n = i >> 7, j = i & 127;
  catpre[(long)n*D1 + 512 + j] = xf[(long)n*(PS+SL) + j];
}
// fused GELU mix + LayerNorm; one block (256 thr) per row of 768
__global__ void fuse_ln_kernel(float* catpre, float* cat,
                               const float* ln_w, const float* ln_b, const float* focus){
  __shared__ float s1[256], s2[256];
  int row = blockIdx.x, j = threadIdx.x;
  float f = focus[0];
  float xp  = catpre[(long)row*D1 + j];
  float xpr = catpre[(long)row*D1 + 512 + j];
  float t = f*xpr + (2.f - f)*xp;
  float g = 0.5f * t * (1.f + erff(t * 0.70710678118654752f));
  catpre[(long)row*D1 + 256 + j] = g;
  __syncthreads();
  float a = 0.f, b2 = 0.f;
#pragma unroll
  for (int i = 0; i < 3; ++i) {
    float v = catpre[(long)row*D1 + j*3 + i];
    a += v; b2 += v*v;
  }
  s1[j] = a; s2[j] = b2;
  __syncthreads();
  for (int st = 128; st > 0; st >>= 1) {
    if (j < st) { s1[j] += s1[j+st]; s2[j] += s2[j+st]; }
    __syncthreads();
  }
  float mu = s1[0] * (1.f/D1);
  float var = s2[0] * (1.f/D1) - mu*mu;
  float rs = rsqrtf(var + 1e-5f);
#pragma unroll
  for (int i = 0; i < 3; ++i) {
    int idx = j*3 + i;
    float v = catpre[(long)row*D1 + idx];
    cat[(long)row*D1 + idx] = (v - mu)*rs*ln_w[idx] + ln_b[idx];
  }
}
// depthwise causal conv (DC=4) + SiLU; xi = xz[..., :di]
__global__ void dwconv_silu_kernel(const float* __restrict__ xz, const float* __restrict__ cw,
                                   const float* __restrict__ cb, float* __restrict__ xc,
                                   int Bn, int L, int di){
  long tot = (long)Bn*L*di;
  long i = (long)blockIdx.x*256 + threadIdx.x;
  if (i >= tot) return;
  int d = (int)(i % di); long r = i / di;
  int l = (int)(r % L); int b = (int)(r / L);
  float acc = cb[d];
#pragma unroll
  for (int j = 0; j < DCN; ++j) {
    int li = l - (DCN-1) + j;
    if (li >= 0) acc += xz[((long)(b*L + li))*(2*di) + d] * cw[d*DCN + j];
  }
  xc[i] = siluf(acc);
}
// selective scan: one thread per (b, d), 16-state register loop over L
__global__ void scan_kernel(const float* __restrict__ delta, const float* __restrict__ dbl,
                            int ldDbl, int dr,
                            const float* __restrict__ xc, const float* __restrict__ xz,
                            const float* __restrict__ A_log, const float* __restrict__ Dp,
                            float* __restrict__ y, int Bn, int L, int di){
  long idx = (long)blockIdx.x*256 + threadIdx.x;
  if (idx >= (long)Bn*di) return;
  int d = (int)(idx % di), b = (int)(idx / di);
  float A[DSN], h[DSN];
#pragma unroll
  for (int s = 0; s < DSN; ++s) { A[s] = -__expf(A_log[(long)d*DSN + s]); h[s] = 0.f; }
  float Dd = Dp[d];
  for (int l = 0; l < L; ++l) {
    long n = (long)b*L + l;
    float dl  = delta[n*di + d];
    float xcv = xc[n*di + d];
    float zv  = xz[n*(2*di) + di + d];
    const float* Bm = dbl + n*ldDbl + dr;
    const float* Cm = Bm + DSN;
    float accv = 0.f;
#pragma unroll
    for (int s = 0; s < DSN; ++s) {
      float dA = __expf(dl * A[s]);
      h[s] = dA*h[s] + dl * Bm[s] * xcv;
      accv += h[s] * Cm[s];
    }
    y[n*di + d] = (accv + xcv*Dd) * siluf(zv);
  }
}
// x3[b,d] = mean_l( x1[(b,l),d] + x2acc[(b,d),l] )
__global__ void combine_mean_kernel(const float* __restrict__ x1, const float* __restrict__ x2acc,
                                    float* __restrict__ x3){
  int i = blockIdx.x*256 + threadIdx.x; // 32*768
  if (i >= BB*D1) return;
  int d = i % D1, b = i / D1;
  float s = 0.f;
  for (int l = 0; l < CC; ++l)
    s += x1[((long)(b*CC + l))*D1 + d] + x2acc[((long)(b*D1 + d))*CC + l];
  x3[i] = s * (1.f/CC);
}
__global__ void reg2_kernel(const float* __restrict__ h, const float* __restrict__ w2,
                            const float* __restrict__ b2, float* __restrict__ out){
  int b = blockIdx.x*32 + threadIdx.x;
  if (b >= BB) return;
  float s = b2[0];
  for (int k = 0; k < D1/2; ++k) s += h[(long)b*(D1/2) + k] * w2[k];
  out[b] = s;
}

// ---------------- host-side Mamba stack driver ----------------
struct MambaW {
  const float *in_w, *conv_w, *conv_b, *xp_w, *dt_w, *dt_b, *A_log, *Dp, *out_w;
};
static void run_stack(float* x, int Bn, int L, int dm, int di, int dr, const MambaW& w,
                      float* xz, float* xc, float* dbl, float* delta, float* ybuf,
                      hipStream_t s)
{
  const int ldDbl = dr + 2*DSN;
  const long N = (long)Bn * L;
  const int EW = 256;
  for (int i = 0; i < 2; ++i) {
    const float* iw  = w.in_w  + (long)i*2*di*dm;
    const float* cw  = w.conv_w+ (long)i*di*DCN;
    const float* cb  = w.conv_b+ (long)i*di;
    const float* xpw = w.xp_w  + (long)i*ldDbl*di;
    const float* dtw = w.dt_w  + (long)i*di*dr;
    const float* dtb = w.dt_b  + (long)i*di;
    const float* al  = w.A_log + (long)i*di*DSN;
    const float* dp  = w.Dp    + (long)i*di;
    const float* ow  = w.out_w + (long)i*dm*di;
    // xz = x @ in_w^T
    gemm_launch(DenseLoader{x, dm}, iw, nullptr, nullptr, xz, (int)N, dm, 2*di, 2*di, 0, s);
    // xc = silu(depthwise_causal_conv(xi) + cb)
    dwconv_silu_kernel<<<dim3((N*di + EW-1)/EW), dim3(EW), 0, s>>>(xz, cw, cb, xc, Bn, L, di);
    // dbl = xc @ xp_w^T
    gemm_launch(DenseLoader{xc, di}, xpw, nullptr, nullptr, dbl, (int)N, di, ldDbl, ldDbl, 0, s);
    // delta = softplus(dt @ dt_w^T + dt_b)   (dt = dbl[:, :dr])
    gemm_launch(DenseLoader{dbl, ldDbl}, dtw, dtb, nullptr, delta, (int)N, dr, di, di, 1, s);
    // scan -> y (includes +xc*D and *silu(z))
    scan_kernel<<<dim3(((long)Bn*di + EW-1)/EW), dim3(EW), 0, s>>>(delta, dbl, ldDbl, dr,
                                                                   xc, xz, al, dp, ybuf, Bn, L, di);
    // x = y @ out_w^T + x   (residual, in place)
    gemm_launch(DenseLoader{ybuf, di}, ow, nullptr, x, x, (int)N, di, dm, dm, 0, s);
  }
}

// ---------------- entry ----------------
extern "C" void kernel_launch(void* const* d_in, const int* in_sizes, int n_in,
                              void* d_out, int out_size, void* d_ws, size_t ws_size,
                              hipStream_t stream)
{
  const float* x_cwt        = (const float*)d_in[0];
  const float* x_features   = (const float*)d_in[1];
  const float* patch_conv_w = (const float*)d_in[2];
  const float* patch_conv_b = (const float*)d_in[3];
  const float* patch_proj_w = (const float*)d_in[4];
  const float* patch_proj_b = (const float*)d_in[5];
  const float* feat_proj_w  = (const float*)d_in[6];
  const float* feat_proj_b  = (const float*)d_in[7];
  const float* ln_w         = (const float*)d_in[8];
  const float* ln_b         = (const float*)d_in[9];
  const float* focus        = (const float*)d_in[10];
  const float* reg_w1       = (const float*)d_in[11];
  const float* reg_b1       = (const float*)d_in[12];
  const float* reg_w2       = (const float*)d_in[13];
  const float* reg_b2       = (const float*)d_in[14];
  MambaW m1{ (const float*)d_in[15], (const float*)d_in[16], (const float*)d_in[17],
             (const float*)d_in[18], (const float*)d_in[19], (const float*)d_in[20],
             (const float*)d_in[21], (const float*)d_in[22], (const float*)d_in[23] };
  MambaW m2{ (const float*)d_in[24], (const float*)d_in[25], (const float*)d_in[26],
             (const float*)d_in[27], (const float*)d_in[28], (const float*)d_in[29],
             (const float*)d_in[30], (const float*)d_in[31], (const float*)d_in[32] };
  float* out = (float*)d_out;

  // workspace bump allocator (floats)
  float* W = (float*)d_ws;
  size_t off = 0;
  auto alloc = [&](size_t n){ float* p = W + off; off += n; return p; };
  const long ROWS1 = (long)BB*CC;      // 1024
  const long ROWS2 = (long)BB*D1;      // 24576
  float* xp_buf  = alloc(2048*32);
  float* catpre  = alloc(ROWS1*D1);
  float* cat     = alloc(ROWS1*D1);
  float* x1      = alloc(ROWS1*D1);
  float* x2acc   = alloc(ROWS2*CC);
  float* catpbuf = alloc(ROWS2*CC);
  float* xcur    = alloc(ROWS2*CC);    // 786432 floats; also fits (1024x768)
  float* xz      = alloc(3145728);     // max(1024*3072, 24576*128)
  float* xc      = alloc(1572864);     // max(1024*1536, 24576*64)
  float* dbl     = alloc(ROWS2*40);    // >= max(1024*80, 24576*34)
  float* delta   = alloc(1572864);
  float* ybuf    = alloc(1572864);
  float* x3      = alloc(BB*D1);
  float* hbuf    = alloc(BB*(D1/2));
  (void)ws_size; (void)in_sizes; (void)n_in; (void)out_size;

  const int EW = 256;
  const long CATN = ROWS1*D1;

  // ---- front end ----
  // xp = conv(x_cwt, patch_conv_w, stride 16) + b  -> xp_buf[(b,ph,pw)][co]
  gemm_launch(PatchLoader{x_cwt}, patch_conv_w, patch_conv_b, nullptr,
              xp_buf, 2048, CC*PP*PP, CC, CC, 0, stream);
  // x_patched -> catpre cols [0,256)
  gemm_launch(PatchProjLoader{xp_buf}, patch_proj_w, patch_proj_b, nullptr,
              catpre, (int)ROWS1, NP, PS, D1, 0, stream);
  // x_proj low half (copy of features) -> cols [512,640)
  copy_feat_kernel<<<dim3((1024*128 + EW-1)/EW), dim3(EW), 0, stream>>>(catpre, x_features);
  // x_proj high half = features[:, PS:] @ feat_proj_w^T + b -> cols [640,768)
  gemm_launch(DenseLoader{x_features + PS, PS+SL}, feat_proj_w, feat_proj_b, nullptr,
              catpre + 640, (int)ROWS1, SL, PS/2, D1, 0, stream);
  // gelu-fuse (cols [256,512)) + LayerNorm -> cat
  fuse_ln_kernel<<<dim3((int)ROWS1), dim3(256), 0, stream>>>(catpre, cat, ln_w, ln_b, focus);

  // ---- m1 stacks: x1 = stack(cat) + stack(flip(cat)) ----
  copy_kernel<<<dim3((CATN+EW-1)/EW), dim3(EW), 0, stream>>>(xcur, cat, CATN);
  run_stack(xcur, BB, CC, D1, 2*D1, 48, m1, xz, xc, dbl, delta, ybuf, stream);
  copy_kernel<<<dim3((CATN+EW-1)/EW), dim3(EW), 0, stream>>>(x1, xcur, CATN);
  flip_seq_kernel<<<dim3((CATN+EW-1)/EW), dim3(EW), 0, stream>>>(xcur, cat, BB, CC, D1);
  run_stack(xcur, BB, CC, D1, 2*D1, 48, m1, xz, xc, dbl, delta, ybuf, stream);
  add_kernel<<<dim3((CATN+EW-1)/EW), dim3(EW), 0, stream>>>(x1, xcur, CATN);

  // ---- m2 stacks on catp = cat^T: x2acc (in catp layout) ----
  transpose_bld_kernel<<<dim3((CATN+EW-1)/EW), dim3(EW), 0, stream>>>(catpbuf, cat, BB, CC, D1);
  copy_kernel<<<dim3((CATN+EW-1)/EW), dim3(EW), 0, stream>>>(xcur, catpbuf, CATN);
  run_stack(xcur, BB, D1, CC, 2*CC, 2, m2, xz, xc, dbl, delta, ybuf, stream);
  copy_kernel<<<dim3((CATN+EW-1)/EW), dim3(EW), 0, stream>>>(x2acc, xcur, CATN);
  flip_seq_kernel<<<dim3((CATN+EW-1)/EW), dim3(EW), 0, stream>>>(xcur, catpbuf, BB, D1, CC);
  run_stack(xcur, BB, D1, CC, 2*CC, 2, m2, xz, xc, dbl, delta, ybuf, stream);
  add_kernel<<<dim3((CATN+EW-1)/EW), dim3(EW), 0, stream>>>(x2acc, xcur, CATN);

  // ---- head ----
  combine_mean_kernel<<<dim3((BB*D1 + EW-1)/EW), dim3(EW), 0, stream>>>(x1, x2acc, x3);
  gemm_launch(DenseLoader{x3, D1}, reg_w1, reg_b1, nullptr, hbuf, BB, D1, D1/2, D1/2, 0, stream);
  reg2_kernel<<<dim3(1), dim3(32), 0, stream>>>(hbuf, reg_w2, reg_b2, out);
}
